// MultiHeadSelfAttentionWithBias_37099927503250
// MI455X (gfx1250) — compile-verified, matrix operand order checked by reading, not run on hardware
//
#include <hip/hip_runtime.h>
#include <hip/hip_bf16.h>

typedef __attribute__((ext_vector_type(16))) _Float16 v16h;
typedef __attribute__((ext_vector_type(8)))  _Float16 v8h;
typedef __attribute__((ext_vector_type(8)))  float    v8f;

#define DEV __device__ __forceinline__

static constexpr int Nb   = 16;
static constexpr int LSEQ = 1024;
static constexpr int DIM  = 512;
static constexpr int NH   = 8;
static constexpr int HD   = 64;

// ---------------------------------------------------------------------------
// WMMA fragment load (ISA 7.12.2 16-bit A/B layout):
//   lane l<16 : row m=l,    K in {0..7, 16..23}
//   lane l>=16: row m=l-16, K in {8..15, 24..31}
// => two contiguous 16-byte (8-half) runs per lane at K=koff and K=koff+16.
// ---------------------------------------------------------------------------
DEV v16h load_frag_rowmajor(const _Float16* __restrict__ base, int ld) {
  const int lane = threadIdx.x & 31;
  const int m    = lane & 15;
  const int koff = (lane >> 4) << 3;            // 0 or 8
  const _Float16* p = base + m * ld + koff;
  v8h lo = *(const v8h*)(p);
  v8h hi = *(const v8h*)(p + 16);
  v16h r;
#pragma unroll
  for (int i = 0; i < 8; ++i) { r[i] = lo[i]; r[i + 8] = hi[i]; }
  return r;
}

DEV v8f wmma_f16(v16h a, v16h b, v8f c) {
  return __builtin_amdgcn_wmma_f32_16x16x32_f16(
      false, a, false, b, (short)0, c, false, false);
}

// Async global -> LDS copy, 16 bytes per lane, tracked by ASYNCcnt.
DEV void async_copy_b128(const _Float16* src, _Float16* lds_dst) {
  unsigned dst = (unsigned)(unsigned long long)lds_dst;   // addr[31:0] = LDS offset
  asm volatile("global_load_async_to_lds_b128 %0, %1, off"
               :: "v"(dst), "v"(src) : "memory");
}

DEV void wait_async_all() {
  asm volatile("s_wait_asynccnt 0" ::: "memory");
}

// ---------------------------------------------------------------------------
// f32 -> f16 conversion (grid-stride)
// ---------------------------------------------------------------------------
__global__ void cvt_f32_to_f16_kernel(const float* __restrict__ src,
                                      _Float16* __restrict__ dst, int n) {
  int i = blockIdx.x * blockDim.x + threadIdx.x;
  int stride = gridDim.x * blockDim.x;
  for (; i < n; i += stride) dst[i] = (_Float16)src[i];
}

// ---------------------------------------------------------------------------
// Shared GEMM tile prefetch: X tile 128x32 halves (two b128/lane),
// W tile 64x32 halves (one b128/lane).  Both padded to ld=40 halves.
// ---------------------------------------------------------------------------
DEV void gemm_prefetch(const _Float16* __restrict__ Abase,
                       const _Float16* __restrict__ Wbase, int k0,
                       _Float16* xs, _Float16* ws) {
  {
    int row = threadIdx.x >> 1;
    int col = (threadIdx.x & 1) * 16;
    const _Float16* src = Abase + (size_t)row * DIM + k0 + col;
    async_copy_b128(src,     xs + row * 40 + col);
    async_copy_b128(src + 8, xs + row * 40 + col + 8);
  }
  {
    int row = threadIdx.x >> 2;
    int col = (threadIdx.x & 3) * 8;
    async_copy_b128(Wbase + (size_t)row * DIM + k0 + col,
                    ws + row * 40 + col);
  }
}

// ---------------------------------------------------------------------------
// QKV projection: out[(n,h,l,hd)] = f16( A(row) . W(row o)^T + bias[o] )
// Block tile 128(M) x 64(N), K-step 32, double-buffered async LDS pipeline.
// ---------------------------------------------------------------------------
__global__ __launch_bounds__(256)
void qkv_gemm_kernel(const _Float16* __restrict__ A,
                     const _Float16* __restrict__ W,
                     const float* __restrict__ bias,
                     _Float16* __restrict__ out) {
  __shared__ __align__(16) _Float16 Xs[2][128 * 40];
  __shared__ __align__(16) _Float16 Ws[2][64 * 40];

  const int wave  = threadIdx.x >> 5;
  const int lane  = threadIdx.x & 31;
  const int mBase = blockIdx.x * 128;
  const int nBase = blockIdx.y * 64;

  const _Float16* Abase = A + (size_t)mBase * DIM;
  const _Float16* Wbase = W + (size_t)nBase * DIM;

  v8f acc[4] = {};

  gemm_prefetch(Abase, Wbase, 0, Xs[0], Ws[0]);
  int buf = 0;
  for (int k0 = 0; k0 < DIM; k0 += 32, buf ^= 1) {
    wait_async_all();
    __syncthreads();                       // tile `buf` published to all waves
    if (k0 + 32 < DIM)
      gemm_prefetch(Abase, Wbase, k0 + 32, Xs[buf ^ 1], Ws[buf ^ 1]);

    v16h a = load_frag_rowmajor(&Xs[buf][(wave * 16) * 40], 40);
#pragma unroll
    for (int t = 0; t < 4; ++t) {
      v16h b = load_frag_rowmajor(&Ws[buf][(t * 16) * 40], 40);
      acc[t] = wmma_f16(a, b, acc[t]);
    }
  }

  const int half = lane >> 4;
  const int ncol = lane & 15;
#pragma unroll
  for (int t = 0; t < 4; ++t) {
    int o  = nBase + t * 16 + ncol;
    float bv = bias[o];
    int h = o >> 6, hd = o & 63;
#pragma unroll
    for (int r = 0; r < 8; ++r) {
      int row = mBase + wave * 16 + r + half * 8;       // flat (n,l)
      int n = row >> 10, l = row & (LSEQ - 1);
      out[((((size_t)n * NH + h) * LSEQ) + l) * HD + hd] =
          (_Float16)(acc[t][r] + bv);
    }
  }
}

// ---------------------------------------------------------------------------
// Output projection: d_out[row, o] = f32( O(row) . Wo(row o)^T + bo[o] )
// ---------------------------------------------------------------------------
__global__ __launch_bounds__(256)
void out_gemm_kernel(const _Float16* __restrict__ A,
                     const _Float16* __restrict__ W,
                     const float* __restrict__ bias,
                     float* __restrict__ out) {
  __shared__ __align__(16) _Float16 Xs[2][128 * 40];
  __shared__ __align__(16) _Float16 Ws[2][64 * 40];

  const int wave  = threadIdx.x >> 5;
  const int lane  = threadIdx.x & 31;
  const int mBase = blockIdx.x * 128;
  const int nBase = blockIdx.y * 64;

  const _Float16* Abase = A + (size_t)mBase * DIM;
  const _Float16* Wbase = W + (size_t)nBase * DIM;

  v8f acc[4] = {};

  gemm_prefetch(Abase, Wbase, 0, Xs[0], Ws[0]);
  int buf = 0;
  for (int k0 = 0; k0 < DIM; k0 += 32, buf ^= 1) {
    wait_async_all();
    __syncthreads();
    if (k0 + 32 < DIM)
      gemm_prefetch(Abase, Wbase, k0 + 32, Xs[buf ^ 1], Ws[buf ^ 1]);

    v16h a = load_frag_rowmajor(&Xs[buf][(wave * 16) * 40], 40);
#pragma unroll
    for (int t = 0; t < 4; ++t) {
      v16h b = load_frag_rowmajor(&Ws[buf][(t * 16) * 40], 40);
      acc[t] = wmma_f16(a, b, acc[t]);
    }
  }

  const int half = lane >> 4;
  const int ncol = lane & 15;
#pragma unroll
  for (int t = 0; t < 4; ++t) {
    int o  = nBase + t * 16 + ncol;
    float bv = bias[o];
#pragma unroll
    for (int r = 0; r < 8; ++r) {
      int row = mBase + wave * 16 + r + half * 8;
      out[(size_t)row * DIM + o] = acc[t][r] + bv;
    }
  }
}

// ---------------------------------------------------------------------------
// Attention tile prefetchers
// ---------------------------------------------------------------------------
DEV void attn_prefetch_K(const _Float16* __restrict__ Kbase, _Float16* ks) {
  int row = threadIdx.x >> 3;                 // key 0..31
  int col = (threadIdx.x & 7) * 8;            // d
  async_copy_b128(Kbase + (size_t)row * HD + col, ks + row * 72 + col);
}

DEV void attn_load_V_transposed(const _Float16* __restrict__ Vbase, _Float16* vt) {
  int row = threadIdx.x >> 3;                 // key 0..31
  int col = (threadIdx.x & 7) * 8;            // d
  v8h vv = *(const v8h*)(Vbase + (size_t)row * HD + col);
#pragma unroll
  for (int i = 0; i < 8; ++i) vt[(col + i) * 40 + row] = vv[i];
}

// ---------------------------------------------------------------------------
// Flash attention. Block = (128 queries, head h, batch n); 8 waves x 16 q.
// Double-buffered 32-key tiles (K async-to-LDS, V VGPR-transposed),
// online softmax, O written f16 [n][L][512].
// ---------------------------------------------------------------------------
__global__ __launch_bounds__(256)
void attention_kernel(const _Float16* __restrict__ Q,   // [N][H][L][HD]
                      const _Float16* __restrict__ K,
                      const _Float16* __restrict__ V,
                      const float* __restrict__ bias,   // [N][L][L]
                      const int* __restrict__ mask,     // [N][L]
                      _Float16* __restrict__ O) {       // [N][L][DIM]
  __shared__ __align__(16) _Float16 Ks[2][32 * 72];
  __shared__ __align__(16) _Float16 Vt[2][64 * 40];
  __shared__ __align__(16) _Float16 Ps[8][16 * 40];

  const int wave = threadIdx.x >> 5;
  const int lane = threadIdx.x & 31;
  const int h    = blockIdx.y;
  const int n    = blockIdx.z;
  const int qBase = blockIdx.x * 128 + wave * 16;

  const size_t headOff = ((size_t)n * NH + h) * LSEQ * HD;

  const _Float16* qptr = Q + headOff + (size_t)qBase * HD;
  v16h aQ0 = load_frag_rowmajor(qptr, HD);
  v16h aQ1 = load_frag_rowmajor(qptr + 32, HD);

  float mrow[8], lrow[8];
  v8f  o[4] = {};
#pragma unroll
  for (int r = 0; r < 8; ++r) { mrow[r] = -1e30f; lrow[r] = 0.f; }

  const int half = lane >> 4;
  const int ncol = lane & 15;

  attn_prefetch_K(K + headOff, Ks[0]);
  attn_load_V_transposed(V + headOff, Vt[0]);

  int buf = 0;
  for (int kb = 0; kb < LSEQ; kb += 32, buf ^= 1) {
    wait_async_all();
    __syncthreads();          // K (async) + V (ds_store) tiles published
    if (kb + 32 < LSEQ) {
      attn_prefetch_K(K + headOff + (size_t)(kb + 32) * HD, Ks[buf ^ 1]);
      attn_load_V_transposed(V + headOff + (size_t)(kb + 32) * HD, Vt[buf ^ 1]);
    }

    // S = Q K^T : 16 queries x 32 keys, two 16-key halves
    v8f s[2];
#pragma unroll
    for (int j = 0; j < 2; ++j) {
      v16h b0 = load_frag_rowmajor(&Ks[buf][(j * 16) * 72], 72);
      v16h b1 = load_frag_rowmajor(&Ks[buf][(j * 16) * 72 + 32], 72);
      v8f c = {};
      c = wmma_f16(aQ0, b0, c);
      c = wmma_f16(aQ1, b1, c);
      s[j] = c;
    }

    // scale + bias + key-padding mask
#pragma unroll
    for (int j = 0; j < 2; ++j) {
      int col = kb + j * 16 + ncol;
      int mk  = mask[n * LSEQ + col];
      const float* bptr =
          bias + ((size_t)n * LSEQ + (qBase + half * 8)) * LSEQ + col;
#pragma unroll
      for (int r = 0; r < 8; ++r) {
        float val = s[j][r] * 0.125f + bptr[(size_t)r * LSEQ];
        s[j][r] = (mk == 0) ? -1.0e9f : val;
      }
    }

    // online softmax per row (rows split across the two 16-lane halves)
    float alpha[8];
#pragma unroll
    for (int r = 0; r < 8; ++r) {
      float v = fmaxf(s[0][r], s[1][r]);
#pragma unroll
      for (int off = 8; off >= 1; off >>= 1)
        v = fmaxf(v, __shfl_xor(v, off, 16));
      float mnew = fmaxf(mrow[r], v);
      alpha[r] = __expf(mrow[r] - mnew);
      mrow[r] = mnew;
      float p0 = __expf(s[0][r] - mnew);
      float p1 = __expf(s[1][r] - mnew);
      float ps = p0 + p1;
#pragma unroll
      for (int off = 8; off >= 1; off >>= 1)
        ps += __shfl_xor(ps, off, 16);
      lrow[r] = lrow[r] * alpha[r] + ps;
      int prow = r + half * 8;
      Ps[wave][prow * 40 + ncol]      = (_Float16)p0;
      Ps[wave][prow * 40 + 16 + ncol] = (_Float16)p1;
#pragma unroll
      for (int t = 0; t < 4; ++t) o[t][r] *= alpha[r];
    }

    // O += P V
    v16h aP = load_frag_rowmajor(&Ps[wave][0], 40);
#pragma unroll
    for (int t = 0; t < 4; ++t) {
      v16h bV = load_frag_rowmajor(&Vt[buf][(t * 16) * 40], 40);
      o[t] = wmma_f16(aP, bV, o[t]);
    }
  }

  // finalize: divide by row sums, write f16 [n][L][h*64 + d]
#pragma unroll
  for (int r = 0; r < 8; ++r) {
    float inv = 1.0f / lrow[r];
    int row = qBase + r + half * 8;
    _Float16* optr = O + ((size_t)n * LSEQ + row) * DIM + h * HD;
#pragma unroll
    for (int t = 0; t < 4; ++t)
      optr[t * 16 + ncol] = (_Float16)(o[t][r] * inv);
  }
}

// ---------------------------------------------------------------------------
// Launch
// ---------------------------------------------------------------------------
extern "C" void kernel_launch(void* const* d_in, const int* in_sizes, int n_in,
                              void* d_out, int out_size, void* d_ws, size_t ws_size,
                              hipStream_t stream) {
  (void)in_sizes; (void)n_in; (void)out_size; (void)ws_size;
  const float* x         = (const float*)d_in[0];
  const float* attn_bias = (const float*)d_in[1];
  const int*   kpm       = (const int*)d_in[2];
  const float* Wq = (const float*)d_in[3];
  const float* bq = (const float*)d_in[4];
  const float* Wk = (const float*)d_in[5];
  const float* bk = (const float*)d_in[6];
  const float* Wv = (const float*)d_in[7];
  const float* bv = (const float*)d_in[8];
  const float* Wo = (const float*)d_in[9];
  const float* bo = (const float*)d_in[10];
  float* out = (float*)d_out;

  const size_t xElems = (size_t)Nb * LSEQ * DIM;   // 8,388,608
  const size_t wElems = (size_t)DIM * DIM;         //   262,144

  char* wsp = (char*)d_ws;
  auto carve = [&](size_t halves) {
    _Float16* p = (_Float16*)wsp;
    wsp += ((halves * sizeof(_Float16) + 255) / 256) * 256;
    return p;
  };
  _Float16* xh  = carve(xElems);
  _Float16* wqh = carve(wElems);
  _Float16* wkh = carve(wElems);
  _Float16* wvh = carve(wElems);
  _Float16* woh = carve(wElems);
  _Float16* Qh  = carve(xElems);
  _Float16* Kh  = carve(xElems);
  _Float16* Vh  = carve(xElems);
  _Float16* Oh  = carve(xElems);

  cvt_f32_to_f16_kernel<<<2048, 256, 0, stream>>>(x,  xh,  (int)xElems);
  cvt_f32_to_f16_kernel<<<256,  256, 0, stream>>>(Wq, wqh, (int)wElems);
  cvt_f32_to_f16_kernel<<<256,  256, 0, stream>>>(Wk, wkh, (int)wElems);
  cvt_f32_to_f16_kernel<<<256,  256, 0, stream>>>(Wv, wvh, (int)wElems);
  cvt_f32_to_f16_kernel<<<256,  256, 0, stream>>>(Wo, woh, (int)wElems);

  dim3 gemmGrid((Nb * LSEQ) / 128, DIM / 64);
  qkv_gemm_kernel<<<gemmGrid, 256, 0, stream>>>(xh, wqh, bq, Qh);
  qkv_gemm_kernel<<<gemmGrid, 256, 0, stream>>>(xh, wkh, bk, Kh);
  qkv_gemm_kernel<<<gemmGrid, 256, 0, stream>>>(xh, wvh, bv, Vh);

  dim3 attnGrid(LSEQ / 128, NH, Nb);
  attention_kernel<<<attnGrid, 256, 0, stream>>>(Qh, Kh, Vh, attn_bias, kpm, Oh);

  out_gemm_kernel<<<gemmGrid, 256, 0, stream>>>(Oh, woh, bo, out);
}
